// ChainCRF_1108101562415
// MI455X (gfx1250) — matrix operand order repaired and checked by reading, not compiled, not run on hardware
//
#include <hip/hip_runtime.h>
#include <hip/hip_bf16.h>
#include <math.h>

typedef __bf16 bf16_t;
typedef bf16_t v16bf __attribute__((ext_vector_type(16)));
typedef bf16_t v8bf  __attribute__((ext_vector_type(8)));
typedef float  v8f   __attribute__((ext_vector_type(8)));

constexpr int Bn = 32, Ln = 256, Cn = 128;
constexpr int MT = 16;   // batches per block

#define LOG2E 1.4426950408889634f
#define LN2   0.6931471805599453f

// native hardware transcendentals: v_exp_f32 (exp2), v_log_f32 (log2)
#define EXP2F(x) __builtin_amdgcn_exp2f(x)
#define LOG2F(x) __builtin_amdgcn_logf(x)

__device__ __forceinline__ bf16_t cvt_bf16(float f) {
  unsigned u = __float_as_uint(f);
  u = (u + 0x7FFFu + ((u >> 16) & 1u)) >> 16;   // round-to-nearest-even
  unsigned short h = (unsigned short)u;
  bf16_t b;
  __builtin_memcpy(&b, &h, 2);
  return b;
}

// 32B global -> LDS, async engine (ASYNCcnt), no VGPR round-trip.
__device__ __forceinline__ void async_copy32(const float* gsrc, unsigned lds_off) {
  asm volatile("global_load_async_to_lds_b128 %0, %1, off"
               :: "v"(lds_off), "v"(gsrc) : "memory");
  asm volatile("global_load_async_to_lds_b128 %0, %1, off offset:16"
               :: "v"(lds_off), "v"(gsrc) : "memory");
}

__device__ __forceinline__ void wait_async0() {
  asm volatile("s_wait_asynccnt 0x0" ::: "memory");
}

__global__ __launch_bounds__(256, 1)
void crf_scan_kernel(const int* __restrict__ tokens, const int* __restrict__ target,
                     const float* __restrict__ mask, const float* __restrict__ stab,
                     const float* __restrict__ trans, float* __restrict__ out) {
  __shared__ __align__(32) bf16_t EtLDS[Cn * Cn];       // exp(trans)^T bf16 (32 KB)
  __shared__ __align__(16) float  partLDS[MT * Cn];     // recurrence state (8 KB)
  __shared__ __align__(16) bf16_t AexpLDS[MT * Cn];     // exp(part-pmax) bf16 (4 KB)
  __shared__ __align__(16) float  emisLDS[2][MT * Cn];  // double-buffered emissions (16 KB)
  __shared__ float pmaxLDS[MT];
  __shared__ float maskLDS[MT];
  __shared__ float blkmaxLDS[MT * 8];                   // per-(row, wave) column-block maxima
  __shared__ float redLDS[MT * 16];
  __shared__ float tgtLDS[MT];

  const int tid  = threadIdx.x;
  const int wave = tid >> 5;         // wave w owns output columns [16w, 16w+16)
  const int lane = tid & 31;
  const int ln   = lane & 15;
  const int hi   = lane >> 4;
  const int j0   = wave * 16;
  const int b0   = blockIdx.x * MT;
  const int em   = tid >> 4;         // staging row (batch) for this thread
  const int ec0  = (tid & 15) * 8;   // staging column chunk (8 floats = 32B)

  const unsigned ldsE0 = (unsigned)(uintptr_t)&emisLDS[0][em * Cn + ec0];
  const unsigned ldsE1 = (unsigned)(uintptr_t)&emisLDS[1][em * Cn + ec0];

  // ---- one-time: Et = exp(trans)^T in bf16 ----
  for (int idx = tid; idx < Cn * Cn; idx += 256) {
    int i = idx >> 7, j = idx & (Cn - 1);
    EtLDS[j * Cn + i] = cvt_bf16(EXP2F(trans[idx] * LOG2E));
  }

  // ---- part0[b,j] = (trans[C-1,j] + emis[b,0,j]) * mask[b,0] ----
  {
    int tok = tokens[(b0 + em) * Ln];
    float mk = mask[(b0 + em) * Ln];
    for (int q = 0; q < 8; ++q)
      partLDS[em * Cn + ec0 + q] =
          (trans[(Cn - 1) * Cn + ec0 + q] + stab[tok * Cn + ec0 + q]) * mk;
  }

  // ---- target-energy: parallel over (batch, 16-step chunk) ----
  {
    int tc = tid & 15;
    float s = 0.f;
    for (int tt = tc * 16; tt < tc * 16 + 16; ++tt) {
      int tg = target[(b0 + em) * Ln + tt];
      int pv = (tt == 0) ? (Cn - 1) : target[(b0 + em) * Ln + tt - 1];
      int tk = tokens[(b0 + em) * Ln + tt];
      s += (trans[pv * Cn + tg] + stab[tk * Cn + tg]) * mask[(b0 + em) * Ln + tt];
    }
    redLDS[em * 16 + tc] = s;
  }

  // prefetch token / mask for step 1
  int   tokA  = tokens[(b0 + em) * Ln + 1];
  float rmask = mask[(b0 + em) * Ln + 1];

  __syncthreads();

  if (tid < MT) {              // reduce target energy
    float a = 0.f;
    for (int k = 0; k < 16; ++k) a += redLDS[tid * 16 + k];
    tgtLDS[tid] = a;
  }
  if (tid < 128) {             // seed blkmax from part0
    int row = tid >> 3, wb = tid & 7;
    float v = partLDS[row * Cn + wb * 16];
    for (int k = 1; k < 16; ++k) v = fmaxf(v, partLDS[row * Cn + wb * 16 + k]);
    blkmaxLDS[row * 8 + wb] = v;
  }

  // ---- preload step-invariant B fragments: B[k][n] = exp(trans[k][j0+n]) ----
  v16bf bfrag[4];
  for (int kc = 0; kc < 4; ++kc)
    bfrag[kc] = *(const v16bf*)&EtLDS[(j0 + ln) * Cn + kc * 32 + hi * 16];

  // kick off async emission gather for t = 1 (buffer 1)
  async_copy32(&stab[tokA * Cn + ec0], ldsE1);
  tokA = tokens[(b0 + em) * Ln + 2];

  // ---- sequential scan over t = 1..L-1 (2 barriers/step) ----
  for (int t = 1; t < Ln; ++t) {
    __syncthreads();                       // A: part/blkmax ready, prev prefetch issued
    wait_async0();                         // emis[t] is in LDS (this wave's copies)
    if (t + 1 < Ln)                        // prefetch emis[t+1] into the other buffer
      async_copy32(&stab[tokA * Cn + ec0], ((t + 1) & 1) ? ldsE1 : ldsE0);
    if ((tid & 15) == 0) maskLDS[em] = rmask;
    rmask = mask[(b0 + em) * Ln + min(t + 1, Ln - 1)];
    tokA  = tokens[(b0 + em) * Ln + min(t + 2, Ln - 1)];

    // pmax[m] from the 8 per-wave block maxima; build A = exp2((part-pmax)*log2e)
    float pm = blkmaxLDS[em * 8];
    for (int w = 1; w < 8; ++w) pm = fmaxf(pm, blkmaxLDS[em * 8 + w]);
    if ((tid & 15) == 0) pmaxLDS[em] = pm;
    v8bf av;
    for (int q = 0; q < 8; ++q)
      av[q] = cvt_bf16(EXP2F((partLDS[em * Cn + ec0 + q] - pm) * LOG2E));
    *(v8bf*)&AexpLDS[em * Cn + ec0] = av;
    __syncthreads();                       // B: Aexp + emis + pmax + mask visible

    // (16x128)x(128x16) tile: 4 x v_wmma_f32_16x16x32_bf16
    v8f acc = {};
    for (int kc = 0; kc < 4; ++kc) {
      v8bf alo = *(const v8bf*)&AexpLDS[ln * Cn + kc * 32 + hi * 16];
      v8bf ahi = *(const v8bf*)&AexpLDS[ln * Cn + kc * 32 + hi * 16 + 8];
      v16bf af;
      for (int e = 0; e < 8; ++e) { af[e] = alo[e]; af[e + 8] = ahi[e]; }
      acc = __builtin_amdgcn_wmma_f32_16x16x32_bf16(
          false, af, false, bfrag[kc], (short)0, acc, false, false);
    }

    // D: VGPR r -> M = r + 8*hi, N = ln. Update state + produce block maxima.
    const float* emis = emisLDS[t & 1];
    const int j = j0 + ln;
    float nv[8];
    for (int r = 0; r < 8; ++r) {
      int m = r + 8 * hi;
      float v = pmaxLDS[m] + LN2 * LOG2F(acc[r]) + emis[m * Cn + j];
      float ov = partLDS[m * Cn + j];
      v = ov + (v - ov) * maskLDS[m];
      partLDS[m * Cn + j] = v;
      nv[r] = v;
    }
    float bm[8];
    for (int r = 0; r < 8; ++r) {
      float v = nv[r];
      for (int off = 8; off; off >>= 1) v = fmaxf(v, __shfl_xor(v, off));
      bm[r] = v;
    }
    if (ln == 0)
      for (int r = 0; r < 8; ++r) blkmaxLDS[(r + 8 * hi) * 8 + wave] = bm[r];
  }

  // ---- final: out[b] = logsumexp_j(part[b,j]) - tgt_energy[b] ----
  __syncthreads();
  {
    int row = 2 * wave + hi;
    float mx = -3.402823466e38f;
    for (int q = 0; q < 8; ++q)
      mx = fmaxf(mx, partLDS[row * Cn + ln + 16 * q]);
    for (int off = 8; off; off >>= 1)
      mx = fmaxf(mx, __shfl_xor(mx, off));
    float sm = 0.f;
    for (int q = 0; q < 8; ++q)
      sm += EXP2F((partLDS[row * Cn + ln + 16 * q] - mx) * LOG2E);
    for (int off = 8; off; off >>= 1)
      sm += __shfl_xor(sm, off);
    if (ln == 0) out[b0 + row] = mx + LN2 * LOG2F(sm) - tgtLDS[row];
  }
}

extern "C" void kernel_launch(void* const* d_in, const int* in_sizes, int n_in,
                              void* d_out, int out_size, void* d_ws, size_t ws_size,
                              hipStream_t stream) {
  const int*   tokens = (const int*)d_in[0];
  const int*   target = (const int*)d_in[1];
  const float* mask   = (const float*)d_in[2];
  const float* stab   = (const float*)d_in[3];
  const float* trans  = (const float*)d_in[4];
  float* out = (float*)d_out;
  crf_scan_kernel<<<Bn / MT, 256, 0, stream>>>(tokens, target, mask, stab, trans, out);
}
